// AFNO2D_31628139168440
// MI455X (gfx1250) — compile-verified, matrix operand order checked by reading, not run on hardware
//
#include <hip/hip_runtime.h>

// AFNO2D on MI455X (gfx1250): all five stages (W-rDFT, H-DFT, block-diag
// complex MLP, inverse H-DFT, inverse W-rDFT) are bf16 WMMA GEMMs.
// DFT/weight matrices are pre-packed into WMMA fragment layout (bf16, scale
// and sign baked in) so every A/B fragment build is two aligned 16B loads.
// Spectrum is stored bf16 in stage-specific layouts so K is always contiguous.
// The inverse-W Nyquist column (cos=+/-1, sin=0) is peeled into a VALU update,
// letting the irfft run as an unguarded K=64 GEMM on a packed (B,H,C,64) buffer.

namespace {
constexpr int kBn = 4, kHn = 128, kWn = 128, kCn = 768, kWf = 65, kNB = 8, kBS = 96;
constexpr float kLam = 0.01f;
constexpr float kScale = 0.08838834764831845f;      // 1/sqrt(128) per-axis ortho
constexpr float kTwoPiOver128 = 0.049087385212340517f;
constexpr size_t kS = (size_t)kBn * kHn * kWf * kCn;       // 25,559,040 spec elems
constexpr size_t kU = 2 * kS;                              // union region elems
constexpr size_t kSC = (size_t)kBn * kHn * kCn * 64;       // 25,165,824 SpecC elems
}

typedef __attribute__((ext_vector_type(16))) __bf16 v16bf;
typedef __attribute__((ext_vector_type(8)))  __bf16 v8bf;
typedef __attribute__((ext_vector_type(8)))  float  v8f;

// K index inside a 16x32 bf16 fragment (CDNA5 ISA 7.12.2):
// lanes 0-15: V0..3 -> K 0..7, V4..7 -> K 16..23 ; lanes 16-31: +8.
__device__ __forceinline__ int kmap(int j, int lane) {
  int v = j >> 1;
  int kb = (v < 4) ? (v << 1) : ((v << 1) + 8);
  return kb + (j & 1) + ((lane & 16) ? 8 : 0);
}

__device__ __forceinline__ v8f wmma_bf16(v16bf a, v16bf b, v8f c) {
  return __builtin_amdgcn_wmma_f32_16x16x32_bf16(false, a, false, b, (short)0, c,
                                                 false, false);
}

// Fragment from a K-contiguous bf16 row: two aligned 16-byte loads.
__device__ __forceinline__ v16bf frag_ld(const __bf16* __restrict__ row, int k0,
                                         int lane) {
  const __bf16* p = row + k0 + ((lane & 16) ? 8 : 0);
  v8bf lo = *(const v8bf*)(p);
  v8bf hi = *(const v8bf*)(p + 16);
  return __builtin_shufflevector(lo, hi, 0, 1, 2, 3, 4, 5, 6, 7, 8, 9, 10, 11, 12,
                                 13, 14, 15);
}

// ------------------------------------------------------------ init: packing
// DFT tables in fragment layout: idx = ((mt*4+kt)*32+lane)*16+j, m=mt*16+lane%16,
// k=kt*32+kmap(j,lane). ACs=cos*s, ASn=-sin*s, ASp=+sin*s.
// irfft tables: 2 ksteps only (K=0..63), Hermitian weight baked (f=0 -> 1, else 2).
__global__ void k_pack_dft(__bf16* __restrict__ ACs, __bf16* __restrict__ ASn,
                           __bf16* __restrict__ ASp, __bf16* __restrict__ AGr,
                           __bf16* __restrict__ AGi) {
  int mt = blockIdx.x, kt = blockIdx.y, lane = threadIdx.x;
  int m = mt * 16 + (lane & 15);
#pragma unroll
  for (int j = 0; j < 16; ++j) {
    int k = kt * 32 + kmap(j, lane);
    float ang = kTwoPiOver128 * (float)((m * k) & 127);
    float s, c;
    __sincosf(ang, &s, &c);
    c *= kScale;
    s *= kScale;
    size_t idx = ((size_t)(mt * 4 + kt) * 32 + lane) * 16 + j;
    ACs[idx] = (__bf16)c;
    ASn[idx] = (__bf16)(-s);
    ASp[idx] = (__bf16)s;
    if (kt < 2) {
      size_t g = ((size_t)(mt * 2 + kt) * 32 + lane) * 16 + j;
      float wgt = (k == 0) ? 1.f : 2.f;
      AGr[g] = (__bf16)(wgt * c);
      AGi[g] = (__bf16)(-wgt * s);
    }
  }
}

// Weights as B-fragments: per table idx = (((blk*3+kt)*6+nt)*32+lane)*16+j,
// value w[k][n]. Tables: W1r, W1i(+), W1i(-), W2r, W2i(+), W2i(-).
__global__ void k_pack_wts(const float* __restrict__ w1, const float* __restrict__ w2,
                           __bf16* __restrict__ P) {
  int blk = blockIdx.x, kt = blockIdx.y, nt = blockIdx.z, lane = threadIdx.x;
  const size_t TB = (size_t)kNB * 3 * 6 * 32 * 16;  // elems per table
  int n = nt * 16 + (lane & 15);
#pragma unroll
  for (int j = 0; j < 16; ++j) {
    int k = kt * 32 + kmap(j, lane);
    size_t idx = (((size_t)(blk * 3 + kt) * 6 + nt) * 32 + lane) * 16 + j;
    float w1r = w1[((size_t)(0 * kNB + blk) * kBS + k) * kBS + n];
    float w1i = w1[((size_t)(1 * kNB + blk) * kBS + k) * kBS + n];
    float w2r = w2[((size_t)(0 * kNB + blk) * kBS + k) * kBS + n];
    float w2i = w2[((size_t)(1 * kNB + blk) * kBS + k) * kBS + n];
    P[0 * TB + idx] = (__bf16)w1r;
    P[1 * TB + idx] = (__bf16)w1i;
    P[2 * TB + idx] = (__bf16)(-w1i);
    P[3 * TB + idx] = (__bf16)w2r;
    P[4 * TB + idx] = (__bf16)w2i;
    P[5 * TB + idx] = (__bf16)(-w2i);
  }
}

// -------------------------------- transpose: x (B,H,W,C) f32 -> xT (B,H,C,W) bf16
__global__ __launch_bounds__(256) void k_transpose(const float* __restrict__ x,
                                                   __bf16* __restrict__ xT) {
  __shared__ float t[32][33];
  int plane = blockIdx.x;                   // b*128+h
  int c0 = blockIdx.y * 32, w0 = blockIdx.z * 32;
  const float* xp = x + (size_t)plane * kWn * kCn;
  __bf16* tp = xT + (size_t)plane * kCn * kWn;
#pragma unroll
  for (int i = threadIdx.y; i < 32; i += 8)
    t[i][threadIdx.x] = xp[(size_t)(w0 + i) * kCn + c0 + threadIdx.x];
  __syncthreads();
#pragma unroll
  for (int i = threadIdx.y; i < 32; i += 8)
    tp[(size_t)(c0 + i) * kWn + w0 + threadIdx.x] = (__bf16)t[threadIdx.x][i];
}

// ----------------- stage 1: rDFT along W. xT (B,H,C,W) -> SpecA (B,Wf,C,H) r/i
__global__ __launch_bounds__(160) void k_dft_w(const __bf16* __restrict__ xT,
                                               __bf16* __restrict__ sar,
                                               __bf16* __restrict__ sai,
                                               const __bf16* __restrict__ ACs,
                                               const __bf16* __restrict__ ASn) {
  int plane = blockIdx.x;                   // b*128+h
  int b = plane >> 7, h = plane & 127;
  int c0 = blockIdx.y * 64;
  int wave = threadIdx.x >> 5, lane = threadIdx.x & 31;
  int n = lane & 15, mhi = (lane & 16) ? 8 : 0;
  const v16bf* Ac = (const v16bf*)ACs;
  const v16bf* As = (const v16bf*)ASn;
  v8f ar[4] = {}, ai[4] = {};
  for (int kt = 0; kt < 4; ++kt) {
    v16bf Ar = Ac[(wave * 4 + kt) * 32 + lane];
    v16bf Ai = As[(wave * 4 + kt) * 32 + lane];
#pragma unroll
    for (int nt = 0; nt < 4; ++nt) {
      const __bf16* row = xT + ((size_t)plane * kCn + c0 + nt * 16 + n) * kWn;
      v16bf Bx = frag_ld(row, kt * 32, lane);
      ar[nt] = wmma_bf16(Ar, Bx, ar[nt]);
      ai[nt] = wmma_bf16(Ai, Bx, ai[nt]);
    }
  }
  // scatter to (B,Wf,C,H): addr = ((b*65+f)*768 + c)*128 + h
#pragma unroll
  for (int r = 0; r < 8; ++r) {
    int f = wave * 16 + r + mhi;
    if (f < kWf) {
      size_t base = ((size_t)(b * kWf + f) * kCn + c0 + n) * kHn + h;
#pragma unroll
      for (int nt = 0; nt < 4; ++nt) {
        sar[base + (size_t)nt * 16 * kHn] = (__bf16)ar[nt][r];
        sai[base + (size_t)nt * 16 * kHn] = (__bf16)ai[nt][r];
      }
    }
  }
}

// Shared accumulation body for the H-DFT (K=h contiguous in SpecA).
// Dr = tC*Br + tSdr*Bi ; Di = tC*Bi + tSdi*Br.
__device__ __forceinline__ void dft_h_body(const __bf16* sar, const __bf16* sai,
                                           const __bf16* tC, const __bf16* tSdr,
                                           const __bf16* tSdi, int bf, int c0,
                                           int wave, int lane, v8f dr[4],
                                           v8f di[4]) {
  int n = lane & 15;
  const v16bf* Ac = (const v16bf*)tC;
  const v16bf* Adr = (const v16bf*)tSdr;
  const v16bf* Adi = (const v16bf*)tSdi;
  for (int kt = 0; kt < 4; ++kt) {
    v16bf Ar = Ac[(wave * 4 + kt) * 32 + lane];
    v16bf Sr = Adr[(wave * 4 + kt) * 32 + lane];
    v16bf Si = Adi[(wave * 4 + kt) * 32 + lane];
#pragma unroll
    for (int nt = 0; nt < 4; ++nt) {
      size_t rowoff = ((size_t)bf * kCn + c0 + nt * 16 + n) * kHn;
      v16bf Br = frag_ld(sar + rowoff, kt * 32, lane);
      v16bf Bi = frag_ld(sai + rowoff, kt * 32, lane);
      dr[nt] = wmma_bf16(Ar, Br, dr[nt]);
      dr[nt] = wmma_bf16(Sr, Bi, dr[nt]);
      di[nt] = wmma_bf16(Ar, Bi, di[nt]);
      di[nt] = wmma_bf16(Si, Br, di[nt]);
    }
  }
}

// ---- stage 2: forward H-DFT. SpecA (B,Wf,C,H) -> SpecB (B,H,Wf,C)
__global__ __launch_bounds__(256) void k_dft_h_fwd(const __bf16* __restrict__ sar,
                                                   const __bf16* __restrict__ sai,
                                                   __bf16* __restrict__ sbr,
                                                   __bf16* __restrict__ sbi,
                                                   const __bf16* __restrict__ tC,
                                                   const __bf16* __restrict__ tSp,
                                                   const __bf16* __restrict__ tSn) {
  int bf = blockIdx.x;                       // b*65+f
  int b = bf / kWf, f = bf % kWf;
  int c0 = blockIdx.y * 64;
  int wave = threadIdx.x >> 5, lane = threadIdx.x & 31;
  int n = lane & 15, mhi = (lane & 16) ? 8 : 0;
  v8f dr[4] = {}, di[4] = {};
  dft_h_body(sar, sai, tC, tSp, tSn, bf, c0, wave, lane, dr, di);
#pragma unroll
  for (int r = 0; r < 8; ++r) {
    int h = wave * 16 + r + mhi;
    size_t base = ((size_t)(b * kHn + h) * kWf + f) * kCn + c0 + n;
#pragma unroll
    for (int nt = 0; nt < 4; ++nt) {
      sbr[base + nt * 16] = (__bf16)dr[nt][r];
      sbi[base + nt * 16] = (__bf16)di[nt][r];
    }
  }
}

// ---- stage 4: inverse H-DFT. SpecA -> SpecC (B,H,C,64) + Nyquist plane (B,H,C)
__global__ __launch_bounds__(256) void k_dft_h_inv(const __bf16* __restrict__ sar,
                                                   const __bf16* __restrict__ sai,
                                                   __bf16* __restrict__ scr,
                                                   __bf16* __restrict__ sci,
                                                   __bf16* __restrict__ nyqr,
                                                   const __bf16* __restrict__ tC,
                                                   const __bf16* __restrict__ tSn,
                                                   const __bf16* __restrict__ tSp) {
  int bf = blockIdx.x;                       // b*65+f
  int b = bf / kWf, f = bf % kWf;
  int c0 = blockIdx.y * 64;
  int wave = threadIdx.x >> 5, lane = threadIdx.x & 31;
  int n = lane & 15, mhi = (lane & 16) ? 8 : 0;
  v8f dr[4] = {}, di[4] = {};
  dft_h_body(sar, sai, tC, tSn, tSp, bf, c0, wave, lane, dr, di);
  if (f < 64) {
    // packed K-contiguous rows: addr = ((b*128+h)*768 + c)*64 + f
#pragma unroll
    for (int r = 0; r < 8; ++r) {
      int h = wave * 16 + r + mhi;
      size_t base = ((size_t)(b * kHn + h) * kCn + c0 + n) * 64 + f;
#pragma unroll
      for (int nt = 0; nt < 4; ++nt) {
        scr[base + (size_t)nt * 16 * 64] = (__bf16)dr[nt][r];
        sci[base + (size_t)nt * 16 * 64] = (__bf16)di[nt][r];
      }
    }
  } else {
    // Nyquist column: imag part never used by the irfft (sin(pi*w)=0) -> drop it
#pragma unroll
    for (int r = 0; r < 8; ++r) {
      int h = wave * 16 + r + mhi;
      size_t base = (size_t)(b * kHn + h) * kCn + c0 + n;
#pragma unroll
      for (int nt = 0; nt < 4; ++nt) nyqr[base + nt * 16] = (__bf16)dr[nt][r];
    }
  }
}

// ---- stage 3: block-diag complex MLP + softshrink. SpecB -> SpecA (B,Wf,C,H)
__global__ __launch_bounds__(128) void k_mix(const __bf16* __restrict__ sbr,
                                             const __bf16* __restrict__ sbi,
                                             __bf16* __restrict__ sar,
                                             __bf16* __restrict__ sai,
                                             const __bf16* __restrict__ P,
                                             const float* __restrict__ b1,
                                             const float* __restrict__ b2) {
  __shared__ __align__(16) __bf16 o1r[4][16][104];  // 16B-aligned rows
  __shared__ __align__(16) __bf16 o1i[4][16][104];
  const size_t TB = (size_t)kNB * 3 * 6 * 32 * 16;
  int blk = blockIdx.y;
  int wave = threadIdx.x >> 5, lane = threadIdx.x & 31;
  int p0 = blockIdx.x * 64 + wave * 16;
  int n = lane & 15, mhi = (lane & 16) ? 8 : 0;
  const v16bf* W1r = (const v16bf*)(P + 0 * TB);
  const v16bf* W1p = (const v16bf*)(P + 1 * TB);
  const v16bf* W1n = (const v16bf*)(P + 2 * TB);
  const v16bf* W2r = (const v16bf*)(P + 3 * TB);
  const v16bf* W2p = (const v16bf*)(P + 4 * TB);
  const v16bf* W2n = (const v16bf*)(P + 5 * TB);
  const float* B1r = b1 + (0 * kNB + blk) * kBS;
  const float* B1i = b1 + (1 * kNB + blk) * kBS;
  const float* B2r = b2 + (0 * kNB + blk) * kBS;
  const float* B2i = b2 + (1 * kNB + blk) * kBS;

  // layer 1: A = activations (K=c contiguous rows of SpecB)
  const __bf16* arow = sbr + (size_t)(p0 + n) * kCn + blk * kBS;
  const __bf16* irow = sbi + (size_t)(p0 + n) * kCn + blk * kBS;
  v16bf Axr[3], Axi[3];
#pragma unroll
  for (int kt = 0; kt < 3; ++kt) {
    Axr[kt] = frag_ld(arow, kt * 32, lane);
    Axi[kt] = frag_ld(irow, kt * 32, lane);
  }
#pragma unroll
  for (int nt = 0; nt < 6; ++nt) {
    v8f orr = {}, oii = {};
#pragma unroll
    for (int kt = 0; kt < 3; ++kt) {
      size_t wi = ((size_t)(blk * 3 + kt) * 6 + nt) * 32 + lane;
      v16bf Br = W1r[wi], Bp = W1p[wi], Bn = W1n[wi];
      orr = wmma_bf16(Axr[kt], Br, orr);   // Xr*W1r - Xi*W1i
      orr = wmma_bf16(Axi[kt], Bn, orr);
      oii = wmma_bf16(Axi[kt], Br, oii);   // Xi*W1r + Xr*W1i
      oii = wmma_bf16(Axr[kt], Bp, oii);
    }
    float vbr = B1r[nt * 16 + n], vbi = B1i[nt * 16 + n];
#pragma unroll
    for (int r = 0; r < 8; ++r) {
      o1r[wave][r + mhi][nt * 16 + n] = (__bf16)fmaxf(orr[r] + vbr, 0.f);
      o1i[wave][r + mhi][nt * 16 + n] = (__bf16)fmaxf(oii[r] + vbi, 0.f);
    }
  }
  __syncthreads();

  // layer 2: A = o1 from LDS (vector ds loads)
  v16bf Ayr[3], Ayi[3];
#pragma unroll
  for (int kt = 0; kt < 3; ++kt) {
    Ayr[kt] = frag_ld(&o1r[wave][n][0], kt * 32, lane);
    Ayi[kt] = frag_ld(&o1i[wave][n][0], kt * 32, lane);
  }
  // output scatter bases into SpecA (B,Wf,C,H)
  size_t base_r[8];
#pragma unroll
  for (int r = 0; r < 8; ++r) {
    int row = p0 + r + mhi;                  // pos = (b*128+h)*65+f
    int b = row / (kHn * kWf), rem = row % (kHn * kWf);
    int h = rem / kWf, f = rem % kWf;
    base_r[r] = ((size_t)(b * kWf + f) * kCn + blk * kBS + n) * kHn + h;
  }
#pragma unroll
  for (int nt = 0; nt < 6; ++nt) {
    v8f orr = {}, oii = {};
#pragma unroll
    for (int kt = 0; kt < 3; ++kt) {
      size_t wi = ((size_t)(blk * 3 + kt) * 6 + nt) * 32 + lane;
      v16bf Br = W2r[wi], Bp = W2p[wi], Bn = W2n[wi];
      orr = wmma_bf16(Ayr[kt], Br, orr);
      orr = wmma_bf16(Ayi[kt], Bn, orr);
      oii = wmma_bf16(Ayi[kt], Br, oii);
      oii = wmma_bf16(Ayr[kt], Bp, oii);
    }
    float vbr = B2r[nt * 16 + n], vbi = B2i[nt * 16 + n];
#pragma unroll
    for (int r = 0; r < 8; ++r) {
      float yr = orr[r] + vbr, yi = oii[r] + vbi;
      yr = (yr > kLam) ? yr - kLam : ((yr < -kLam) ? yr + kLam : 0.f);
      yi = (yi > kLam) ? yi - kLam : ((yi < -kLam) ? yi + kLam : 0.f);
      size_t a = base_r[r] + (size_t)nt * 16 * kHn;
      sar[a] = (__bf16)yr;
      sai[a] = (__bf16)yi;
    }
  }
}

// ------ stage 5: inverse rDFT along W (K=64 GEMM + Nyquist VALU) + residual
__global__ __launch_bounds__(256) void k_idft_w(const __bf16* __restrict__ scr,
                                                const __bf16* __restrict__ sci,
                                                const __bf16* __restrict__ nyqr,
                                                const float* __restrict__ x,
                                                float* __restrict__ out,
                                                const __bf16* __restrict__ AGr,
                                                const __bf16* __restrict__ AGi) {
  int plane = blockIdx.x;                    // b*128+h
  int c0 = blockIdx.y * 64;
  int wave = threadIdx.x >> 5, lane = threadIdx.x & 31;
  int n = lane & 15, mhi = (lane & 16) ? 8 : 0;
  const v16bf* Gr = (const v16bf*)AGr;
  const v16bf* Gi = (const v16bf*)AGi;
  v8f acc[4] = {};
  for (int kt = 0; kt < 2; ++kt) {           // K=64, unguarded vector loads
    v16bf Ar = Gr[(wave * 2 + kt) * 32 + lane];
    v16bf Ai = Gi[(wave * 2 + kt) * 32 + lane];
#pragma unroll
    for (int nt = 0; nt < 4; ++nt) {
      const __bf16* rr = scr + ((size_t)plane * kCn + c0 + nt * 16 + n) * 64;
      const __bf16* ri = sci + ((size_t)plane * kCn + c0 + nt * 16 + n) * 64;
      acc[nt] = wmma_bf16(Ar, frag_ld(rr, kt * 32, lane), acc[nt]);
      acc[nt] = wmma_bf16(Ai, frag_ld(ri, kt * 32, lane), acc[nt]);
    }
  }
  const float* xp = x + (size_t)plane * kWn * kCn;
  float* op = out + (size_t)plane * kWn * kCn;
#pragma unroll
  for (int nt = 0; nt < 4; ++nt) {
    // Nyquist term: y[w] += (-1)^w * s * Xr_nyq[c]
    float xn = (float)nyqr[(size_t)plane * kCn + c0 + nt * 16 + n] * kScale;
#pragma unroll
    for (int r = 0; r < 8; ++r) {
      int w = wave * 16 + r + mhi;
      float y = acc[nt][r] + ((w & 1) ? -xn : xn);
      size_t idx = (size_t)w * kCn + c0 + nt * 16 + n;
      op[idx] = y + xp[idx];                 // residual
    }
  }
}

extern "C" void kernel_launch(void* const* d_in, const int* in_sizes, int n_in,
                              void* d_out, int out_size, void* d_ws, size_t ws_size,
                              hipStream_t stream) {
  (void)in_sizes; (void)n_in; (void)out_size; (void)ws_size;
  const float* x  = (const float*)d_in[0];
  const float* w1 = (const float*)d_in[1];
  const float* b1 = (const float*)d_in[2];
  const float* w2 = (const float*)d_in[3];
  const float* b2 = (const float*)d_in[4];
  float* out = (float*)d_out;

  __bf16* base = (__bf16*)d_ws;
  // region1 (kU elems) time-multiplexed: xT -> SpecB r+i -> SpecC r+i + Nyquist
  __bf16* xT   = base;
  __bf16* sbr  = base;
  __bf16* sbi  = base + kS;
  __bf16* scr  = base;
  __bf16* sci  = base + kSC;
  __bf16* nyqr = base + 2 * kSC;       // 0.39M elems, fits under kU
  __bf16* sar  = base + kU;
  __bf16* sai  = sar + kS;
  __bf16* ACs  = sai + kS;             // 8*4*32*16 = 16384 elems each
  __bf16* ASn  = ACs + 16384;
  __bf16* ASp  = ASn + 16384;
  __bf16* AGr  = ASp + 16384;          // 8*2*32*16 = 8192 elems each
  __bf16* AGi  = AGr + 8192;
  __bf16* Wpk  = AGi + 8192;           // 6 * 73728 elems

  k_pack_dft<<<dim3(8, 4), dim3(32), 0, stream>>>(ACs, ASn, ASp, AGr, AGi);
  k_pack_wts<<<dim3(8, 3, 6), dim3(32), 0, stream>>>(w1, w2, Wpk);
  k_transpose<<<dim3(kBn * kHn, kCn / 32, kWn / 32), dim3(32, 8), 0, stream>>>(x, xT);
  // W-rDFT: xT -> SpecA
  k_dft_w<<<dim3(kBn * kHn, kCn / 64), dim3(160), 0, stream>>>(xT, sar, sai, ACs,
                                                               ASn);
  // forward H-DFT: SpecA -> SpecB   (Dr: +sin for Bi ; Di: -sin for Br)
  k_dft_h_fwd<<<dim3(kBn * kWf, kCn / 64), dim3(256), 0, stream>>>(
      sar, sai, sbr, sbi, ACs, ASp, ASn);
  // mixing: SpecB -> SpecA
  k_mix<<<dim3((int)((kBn * kHn * kWf) / 64), kNB), dim3(128), 0, stream>>>(
      sbr, sbi, sar, sai, Wpk, b1, b2);
  // inverse H-DFT: SpecA -> SpecC + Nyquist   (Dr: -sin for Bi ; Di: +sin for Br)
  k_dft_h_inv<<<dim3(kBn * kWf, kCn / 64), dim3(256), 0, stream>>>(
      sar, sai, scr, sci, nyqr, ACs, ASn, ASp);
  // inverse W-rDFT + Nyquist + residual: SpecC -> out
  k_idft_w<<<dim3(kBn * kHn, kCn / 64), dim3(256), 0, stream>>>(scr, sci, nyqr, x,
                                                                out, AGr, AGi);
}
